// DeformableAttention_12902081757412
// MI455X (gfx1250) — matrix operand order repaired and checked by reading, not compiled, not run on hardware
//
#include <hip/hip_runtime.h>

// Problem constants (match reference)
#define BB     2
#define CCH    256
#define HH     64
#define WWD    64
#define HWSZ   (HH * WWD)           // 4096
#define MTOT   (BB * HWSZ)          // 8192 pixels
#define KDIM   CCH                  // 256
#define NHP    32                   // heads*points
#define OFFCH  (2 * NHP)            // 64
#define ATTCH  NHP                  // 32

typedef unsigned short u16;
typedef __attribute__((ext_vector_type(16))) __bf16 v16bf;
typedef __attribute__((ext_vector_type(8)))  float  v8f;

union ABPack { struct { uint4 lo, hi; } u; v16bf v; };

__device__ __forceinline__ u16 f2bf(float f) {
    unsigned int u = __float_as_uint(f);
    u += 0x7FFFu + ((u >> 16) & 1u);   // round-to-nearest-even
    return (u16)(u >> 16);
}

// ---------------------------------------------------------------------------
// Convert x [B,C,HW] f32  ->  xT [B*HW, C] bf16 (K-contiguous A matrix)
// ---------------------------------------------------------------------------
__global__ void k_conv_x(const float* __restrict__ x, u16* __restrict__ xt) {
    int idx = blockIdx.x * blockDim.x + threadIdx.x;
    if (idx >= BB * CCH * HWSZ) return;
    int b  = idx / (CCH * HWSZ);
    int r  = idx % (CCH * HWSZ);
    int c  = r / HWSZ;
    int hw = r % HWSZ;
    xt[(size_t)(b * HWSZ + hw) * CCH + c] = f2bf(x[idx]);
}

// Plain f32 -> bf16 copy (weights, already [O,C] K-contiguous)
__global__ void k_conv_w(const float* __restrict__ w, u16* __restrict__ wb, int n) {
    int idx = blockIdx.x * blockDim.x + threadIdx.x;
    if (idx < n) wb[idx] = f2bf(w[idx]);
}

// ---------------------------------------------------------------------------
// Small GEMM (narrow N: off 64, att 32): one wave -> one 16x16 tile.
//   mode 0: store f32 [Mtot, Nout]      mode 1: sigmoid + store
// ---------------------------------------------------------------------------
__global__ void k_gemm_wmma(const u16* __restrict__ A, const u16* __restrict__ Bw,
                            const float* __restrict__ bias, float* __restrict__ out,
                            int Mtot, int Nout, int mode) {
    const int lane = threadIdx.x & 31;
    const int wave = threadIdx.x >> 5;
    const int gw   = blockIdx.x * (blockDim.x >> 5) + wave;
    const int ntiles = Nout >> 4;
    const int mt = gw / ntiles;
    const int nt = gw % ntiles;
    if (mt * 16 >= Mtot) return;               // wave-uniform: EXEC stays all-ones

    const int half = lane >> 4;
    const int l15  = lane & 15;

    const u16* arow = A  + (size_t)(mt * 16 + l15) * KDIM + half * 8;
    const u16* brow = Bw + (size_t)(nt * 16 + l15) * KDIM + half * 8;

    v8f acc = {};
#pragma unroll
    for (int kk = 0; kk < KDIM; kk += 32) {
        ABPack a, b;
        a.u.lo = *(const uint4*)(arow + kk);        // k = g+0..7
        a.u.hi = *(const uint4*)(arow + kk + 16);   // k = g+16..23
        b.u.lo = *(const uint4*)(brow + kk);        // k = g+0..7
        b.u.hi = *(const uint4*)(brow + kk + 16);   // k = g+16..23
        __builtin_prefetch(arow + kk + 64, 0, 1);
        acc = __builtin_amdgcn_wmma_f32_16x16x32_bf16(
            false, a.v, false, b.v, (short)0, acc, false, false);
    }

    const int nCol = nt * 16 + l15;
    const float bv = bias[nCol];
#pragma unroll
    for (int r = 0; r < 8; ++r) {
        const int mrow = mt * 16 + r + half * 8;
        float v = acc[r] + bv;
        if (mode == 1) v = 1.0f / (1.0f + expf(-v));
        out[(size_t)mrow * Nout + nCol] = v;
    }
}

// ---------------------------------------------------------------------------
// Big GEMM (N multiple of 64): block = 256M x 64N panel, wave = 32M x 64N.
// B panel (64 x 256 bf16 = 32 KB) is staged into LDS once per block via the
// CDNA5 async copy engine (global_load_async_to_lds_b128, ASYNCcnt), then
// WMMA's B operands come from LDS (ds_load_b128) while A streams from global.
//   mode 0: store f32 [MTOT, Nout]   (val)
//   mode 2: + resid[b,n,hw], store f32 [B,C,HW]   (final out)
// ---------------------------------------------------------------------------
__global__ void k_gemm_big(const u16* __restrict__ A, const u16* __restrict__ Bw,
                           const float* __restrict__ bias, float* __restrict__ out,
                           const float* __restrict__ resid, int Nout, int mode) {
    __shared__ u16 Bs[64 * KDIM];                 // 32 KB weight panel

    const int lane = threadIdx.x & 31;
    const int wave = threadIdx.x >> 5;
    const int nPanels = Nout >> 6;
    const int mPanel  = blockIdx.x / nPanels;     // 256-row M panel
    const int nPanel  = blockIdx.x % nPanels;     // 64-col N panel

    // ---- async-stage B panel into LDS (2048 x 16B chunks, 8 per thread) ----
    {
        const u16* bsrc = Bw + (size_t)(nPanel * 64) * KDIM;
        unsigned ldsBase = (unsigned)(size_t)&Bs[0];
#pragma unroll
        for (int i = 0; i < 8; ++i) {
            int chunk = threadIdx.x + i * 256;
            unsigned dst = ldsBase + chunk * 16;
            unsigned long long src = (unsigned long long)(size_t)(bsrc + chunk * 8);
            asm volatile("global_load_async_to_lds_b128 %0, %1, off"
                         :: "v"(dst), "v"(src) : "memory");
        }
        asm volatile("s_wait_asynccnt 0x0" ::: "memory");
    }
    __syncthreads();

    const int half = lane >> 4;
    const int l15  = lane & 15;
    const int m0   = (mPanel * 8 + wave) * 32;    // this wave's 32 M rows

    // Per-lane operand base pointers
    const u16* arow0 = A + (size_t)(m0 + l15)      * KDIM + half * 8;
    const u16* arow1 = A + (size_t)(m0 + 16 + l15) * KDIM + half * 8;
    const u16* brow[4];
#pragma unroll
    for (int ni = 0; ni < 4; ++ni)
        brow[ni] = &Bs[(ni * 16 + l15) * KDIM + half * 8];

    v8f acc[8] = {};                               // [mi][ni] -> acc[mi*4+ni]
#pragma unroll
    for (int kk = 0; kk < KDIM; kk += 32) {
        ABPack a0, a1;
        a0.u.lo = *(const uint4*)(arow0 + kk);
        a0.u.hi = *(const uint4*)(arow0 + kk + 16);
        a1.u.lo = *(const uint4*)(arow1 + kk);
        a1.u.hi = *(const uint4*)(arow1 + kk + 16);
        __builtin_prefetch(arow0 + kk + 64, 0, 1);
        __builtin_prefetch(arow1 + kk + 64, 0, 1);
#pragma unroll
        for (int ni = 0; ni < 4; ++ni) {
            ABPack b;
            b.u.lo = *(const uint4*)(brow[ni] + kk);    // LDS: ds_load_b128
            b.u.hi = *(const uint4*)(brow[ni] + kk + 16);
            acc[0 * 4 + ni] = __builtin_amdgcn_wmma_f32_16x16x32_bf16(
                false, a0.v, false, b.v, (short)0, acc[0 * 4 + ni], false, false);
            acc[1 * 4 + ni] = __builtin_amdgcn_wmma_f32_16x16x32_bf16(
                false, a1.v, false, b.v, (short)0, acc[1 * 4 + ni], false, false);
        }
    }

    // ---- epilogue ----
#pragma unroll
    for (int mi = 0; mi < 2; ++mi) {
#pragma unroll
        for (int ni = 0; ni < 4; ++ni) {
            const int nCol = nPanel * 64 + ni * 16 + l15;
            const float bv = bias[nCol];
            const v8f c = acc[mi * 4 + ni];
#pragma unroll
            for (int r = 0; r < 8; ++r) {
                const int mrow = m0 + mi * 16 + r + half * 8;
                float v = c[r] + bv;
                if (mode == 2) {
                    const int b2 = mrow / HWSZ, hw = mrow % HWSZ;
                    const size_t xi = (size_t)b2 * CCH * HWSZ + (size_t)nCol * HWSZ + hw;
                    out[xi] = v + resid[xi];
                } else {
                    out[(size_t)mrow * Nout + nCol] = v;
                }
            }
        }
    }
}

// ---------------------------------------------------------------------------
// Deformable sampling + head aggregation (val resident in 192 MB L2).
// One wave per pixel; lane owns 8 channels; corners are contiguous 1 KB rows.
// ---------------------------------------------------------------------------
__global__ void k_sample(const float* __restrict__ val, const float* __restrict__ off,
                         const float* __restrict__ att, u16* __restrict__ aggb) {
    const int lane = threadIdx.x & 31;
    const int wave = threadIdx.x >> 5;
    const int p    = blockIdx.x * (blockDim.x >> 5) + wave;
    if (p >= MTOT) return;
    const int b  = p / HWSZ, hw = p % HWSZ;
    const int px = hw % WWD, py = hw / WWD;
    const float* valb = val + (size_t)b * HWSZ * CCH;
    const float* offp = off + (size_t)p * OFFCH;
    const float* attp = att + (size_t)p * ATTCH;
    const int cb = lane * 8;

    float acc[8];
#pragma unroll
    for (int r = 0; r < 8; ++r) acc[r] = 0.0f;

    for (int hp = 0; hp < NHP; ++hp) {
        // align_corners=True grid math collapses to ix = x + off_x, iy = y + off_y
        const float ix = (float)px + offp[2 * hp + 0];
        const float iy = (float)py + offp[2 * hp + 1];
        const float a  = attp[hp];
        const float fx = floorf(ix), fy = floorf(iy);
        const float tx = ix - fx,    ty = iy - fy;
        const int   x0 = (int)fx,    y0 = (int)fy;
        const float wc[4] = { (1.f - tx) * (1.f - ty), tx * (1.f - ty),
                              (1.f - tx) * ty,         tx * ty };
        const int xs[4] = { x0, x0 + 1, x0,     x0 + 1 };
        const int ys[4] = { y0, y0,     y0 + 1, y0 + 1 };
#pragma unroll
        for (int c4 = 0; c4 < 4; ++c4) {
            const int xc = xs[c4], yc = ys[c4];
            const bool valid = (xc >= 0) & (xc < WWD) & (yc >= 0) & (yc < HH);
            const float w = valid ? wc[c4] * a : 0.0f;
            const int xcl = min(max(xc, 0), WWD - 1);
            const int ycl = min(max(yc, 0), HH - 1);
            const float* row = valb + (size_t)(ycl * WWD + xcl) * CCH + cb;
            const float4 g0 = *(const float4*)(row);
            const float4 g1 = *(const float4*)(row + 4);
            acc[0] += w * g0.x; acc[1] += w * g0.y; acc[2] += w * g0.z; acc[3] += w * g0.w;
            acc[4] += w * g1.x; acc[5] += w * g1.y; acc[6] += w * g1.z; acc[7] += w * g1.w;
        }
    }

    u16 o[8];
#pragma unroll
    for (int r = 0; r < 8; ++r) o[r] = f2bf(acc[r]);
    *(uint4*)(aggb + (size_t)p * CCH + cb) = *(const uint4*)o;
}

// ---------------------------------------------------------------------------
extern "C" void kernel_launch(void* const* d_in, const int* in_sizes, int n_in,
                              void* d_out, int out_size, void* d_ws, size_t ws_size,
                              hipStream_t stream) {
    const float* x     = (const float*)d_in[0];
    const float* w_off = (const float*)d_in[1];
    const float* b_off = (const float*)d_in[2];
    const float* w_att = (const float*)d_in[3];
    const float* b_att = (const float*)d_in[4];
    const float* w_val = (const float*)d_in[5];
    const float* b_val = (const float*)d_in[6];
    const float* w_out = (const float*)d_in[7];
    const float* b_out = (const float*)d_in[8];
    float* out = (float*)d_out;

    // Workspace carve-up (256 B aligned)
    char* ws = (char*)d_ws;
    size_t o = 0;
    auto alloc = [&](size_t bytes) { char* p = ws + o; o += (bytes + 255) & ~(size_t)255; return p; };
    u16*   xTb  = (u16*)  alloc((size_t)MTOT * KDIM * 2);   // x transposed, bf16
    u16*   aggb = (u16*)  alloc((size_t)MTOT * KDIM * 2);   // aggregated, bf16
    u16*   wvb  = (u16*)  alloc((size_t)CCH  * KDIM * 2);   // w_val bf16
    u16*   wob  = (u16*)  alloc((size_t)CCH  * KDIM * 2);   // w_out bf16
    u16*   wfb  = (u16*)  alloc((size_t)OFFCH* KDIM * 2);   // w_off bf16
    u16*   wab  = (u16*)  alloc((size_t)ATTCH* KDIM * 2);   // w_att bf16
    float* valF = (float*)alloc((size_t)MTOT * CCH  * 4);   // val [M,256]
    float* offF = (float*)alloc((size_t)MTOT * OFFCH* 4);   // off [M,64]
    float* attF = (float*)alloc((size_t)MTOT * ATTCH* 4);   // att [M,32]
    (void)ws_size; (void)n_in; (void)in_sizes; (void)out_size;

    const int T = 256;

    // Stage 0: precision/layout conversion
    k_conv_x<<<(BB * CCH * HWSZ + T - 1) / T, T, 0, stream>>>(x, xTb);
    k_conv_w<<<(CCH  * KDIM + T - 1) / T, T, 0, stream>>>(w_val, wvb, CCH  * KDIM);
    k_conv_w<<<(CCH  * KDIM + T - 1) / T, T, 0, stream>>>(w_out, wob, CCH  * KDIM);
    k_conv_w<<<(OFFCH* KDIM + T - 1) / T, T, 0, stream>>>(w_off, wfb, OFFCH* KDIM);
    k_conv_w<<<(ATTCH* KDIM + T - 1) / T, T, 0, stream>>>(w_att, wab, ATTCH* KDIM);

    // Stage 1: projection GEMMs (WMMA bf16 -> f32 accum)
    // val: big kernel, LDS-staged weights; off/att: small kernel
    k_gemm_big<<<(MTOT / 256) * (CCH / 64), T, 0, stream>>>(xTb, wvb, b_val, valF, nullptr, CCH, 0);
    k_gemm_wmma<<<((MTOT / 16) * (OFFCH / 16) + 7) / 8, T, 0, stream>>>(xTb, wfb, b_off, offF, MTOT, OFFCH, 0);
    k_gemm_wmma<<<((MTOT / 16) * (ATTCH / 16) + 7) / 8, T, 0, stream>>>(xTb, wab, b_att, attF, MTOT, ATTCH, 1);

    // Stage 2: bilinear sampling + attention-weighted head aggregation
    k_sample<<<(MTOT + 7) / 8, T, 0, stream>>>(valF, offF, attF, aggb);

    // Stage 3: output projection + residual (channel-major store)
    k_gemm_big<<<(MTOT / 256) * (CCH / 64), T, 0, stream>>>(aggb, wob, b_out, out, x, CCH, 2);
}
